// MultiHeadAttention_67130338836902
// MI455X (gfx1250) — matrix operand order repaired and checked by reading, not compiled, not run on hardware
//
#include <hip/hip_runtime.h>

typedef _Float16 v4h  __attribute__((ext_vector_type(4)));
typedef _Float16 v8h  __attribute__((ext_vector_type(8)));
typedef _Float16 v16h __attribute__((ext_vector_type(16)));
typedef float    v8f  __attribute__((ext_vector_type(8)));
typedef unsigned int u32x4 __attribute__((ext_vector_type(4)));
typedef int      i32x4 __attribute__((ext_vector_type(4)));
typedef int      i32x8 __attribute__((ext_vector_type(8)));

#ifndef MHA_HAVE_TDM
#if defined(__has_builtin)
#if __has_builtin(__builtin_amdgcn_tensor_load_to_lds) && \
    __has_builtin(__builtin_amdgcn_s_wait_tensorcnt)
#define MHA_HAVE_TDM 1
#endif
#endif
#endif
#ifndef MHA_HAVE_TDM
#define MHA_HAVE_TDM 0
#endif

__device__ __forceinline__ v16h cat16(v8h a, v8h b) {
  return __builtin_shufflevector(a, b, 0,1,2,3,4,5,6,7,8,9,10,11,12,13,14,15);
}

__device__ __forceinline__ v8f wmma_f16(v16h a, v16h b, v8f c) {
  // D = A(16x32 f16) * B(32x16 f16) + C(16x16 f32)
  return __builtin_amdgcn_wmma_f32_16x16x32_f16(false, a, false, b, (short)0, c,
                                                false, false);
}

#if MHA_HAVE_TDM
// Build a D# for a 2D f16 tile (tile_d1 rows x tile_d0 elems) of a row-major
// tensor (tensor_d1 rows x tensor_d0 elems, row stride stride0 elems) and
// issue TENSOR_LOAD_TO_LDS. Packed row-major into LDS at lds_addr.
// D# layout per CDNA5 ISA.
// 6-arg builtin form (clang-23 / therock-10.0 toolchain).
__device__ __forceinline__ void tdm_load_2d_f16(
    unsigned lds_addr, const void* gptr, unsigned tensor_d0, unsigned tensor_d1,
    unsigned tile_d0, unsigned tile_d1, unsigned stride0) {
  const unsigned long long ga = (unsigned long long)gptr;
  u32x4 g0;
  g0[0] = 1u;                                    // count=1, user desc
  g0[1] = lds_addr;                              // bits 63:32 lds_addr
  g0[2] = (unsigned)ga;                          // global_addr[31:0]
  g0[3] = (unsigned)(ga >> 32) | (2u << 30);     // global_addr[56:32] | type=2
  i32x8 g1;
  g1[0] = 0x00010000;                            // data_size=1 (2 bytes)
  g1[1] = (int)((tensor_d0 & 0xffffu) << 16);    // tensor_dim0[15:0]
  g1[2] = (int)((tensor_d0 >> 16) | ((tensor_d1 & 0xffffu) << 16));
  g1[3] = (int)((tensor_d1 >> 16) | (tile_d0 << 16));   // tile_dim0
  g1[4] = (int)tile_d1;                          // tile_dim1 (tile_dim2=0)
  g1[5] = (int)stride0;                          // tensor_dim0_stride[31:0]
  g1[6] = 0;
  g1[7] = 0;
  const i32x4 z4 = {0, 0, 0, 0};
  const i32x8 z8 = {0, 0, 0, 0, 0, 0, 0, 0};
  __builtin_amdgcn_tensor_load_to_lds(g0, g1, z4, z4, z8, 0);
}
#endif

// ---------------------------------------------------------------------------
// fp32 -> f16 conversion (vectorized x4)
// ---------------------------------------------------------------------------
__global__ void mha_cvt_f32_to_f16(const float* __restrict__ in,
                                   _Float16* __restrict__ out, int n4) {
  int i = blockIdx.x * blockDim.x + threadIdx.x;
  if (i < n4) {
    const float4 v = ((const float4*)in)[i];
    v4h o = { (_Float16)v.x, (_Float16)v.y, (_Float16)v.z, (_Float16)v.w };
    ((v4h*)out)[i] = o;
  }
}

// ---------------------------------------------------------------------------
// WMMA GEMM: C[M,N] = A[M,K] * B[K,N] + bias
// MODE 0: write f16 in head layout [B,H,S,64]    (QKV projections)
// MODE 1: write f32 row-major [M,N] to d_out     (output projection)
// Tile: 128x64, 8 waves (4x2), each wave 32x32 (2x2 WMMA frags), K-step 32.
// A tile staged by the Tensor Data Mover; B tile transposed manually.
// ---------------------------------------------------------------------------
template <int MODE>
__global__ __launch_bounds__(256) void mha_gemm_wmma(
    const _Float16* __restrict__ A, const _Float16* __restrict__ Bw,
    const float* __restrict__ bias, _Float16* __restrict__ outh,
    float* __restrict__ outf, int M, int N, int K, int S) {
  __shared__ _Float16 sA[128][32];   // A tile, row-major
  __shared__ _Float16 sBt[64][32];   // B tile, transposed (n, k)

  const int tid  = threadIdx.x;
  const int wave = tid >> 5, lane = tid & 31;
  const int wm = wave >> 1, wn = wave & 1;
  const int lg = lane >> 4, lr = lane & 15;
  const int tm = blockIdx.x * 128, tn = blockIdx.y * 64;
#if MHA_HAVE_TDM
  const unsigned sA_lds = (unsigned)(unsigned long long)(void*)&sA[0][0];
#endif

  v8f acc[2][2] = {};

  for (int k0 = 0; k0 < K; k0 += 32) {
#if MHA_HAVE_TDM
    // DMA the 128x32 A tile into LDS (wave 0 issues; TENSORcnt tracks it)
    if (wave == 0)
      tdm_load_2d_f16(sA_lds, A + (size_t)tm * K + k0, K, M, 32, 128, K);
#else
    {  // stage A tile: 128 rows x 32 halves; each thread moves 32 bytes
      const int row = tid >> 1, seg = tid & 1;
      const v8h* src = (const v8h*)(A + (size_t)(tm + row) * K + k0 + seg * 16);
      v8h* dst = (v8h*)&sA[row][seg * 16];
      dst[0] = src[0];
      dst[1] = src[1];
    }
#endif
    {  // stage B tile transposed: global row-major [K,N] -> LDS [n][k]
      const int kk = tid >> 3, ns = tid & 7;
      const v8h tmp = *(const v8h*)(Bw + (size_t)(k0 + kk) * N + tn + ns * 8);
#pragma unroll
      for (int j = 0; j < 8; ++j) sBt[ns * 8 + j][kk] = tmp[j];
      if (k0 + 32 < K)
        __builtin_prefetch(Bw + (size_t)(k0 + 32 + kk) * N + tn + ns * 8, 0, 1);
    }
#if MHA_HAVE_TDM
    if (wave == 0) __builtin_amdgcn_s_wait_tensorcnt(0);
#endif
    __syncthreads();

#pragma unroll
    for (int fm = 0; fm < 2; ++fm) {
      // A frag (16x32): lane row = lr, halves j -> K = lg*8+j, 8+j -> 16+lg*8+j
      const v8h* pa = (const v8h*)&sA[wm * 32 + fm * 16 + lr][0];
      const v16h af = cat16(pa[lg], pa[2 + lg]);
#pragma unroll
      for (int fn = 0; fn < 2; ++fn) {
        // B frag (32x16): lane col = lr, halves j -> K = lg*8+j, 8+j -> 16+lg*8+j
        const v8h* pb = (const v8h*)&sBt[wn * 32 + fn * 16 + lr][0];
        const v16h bf = cat16(pb[lg], pb[2 + lg]);
        acc[fm][fn] = wmma_f16(af, bf, acc[fm][fn]);
      }
    }
    __syncthreads();
  }

  // C/D layout: VGPR r, lane l: M = (l/16)*8 + r, N = l%16
#pragma unroll
  for (int fm = 0; fm < 2; ++fm)
#pragma unroll
    for (int fn = 0; fn < 2; ++fn)
#pragma unroll
      for (int r = 0; r < 8; ++r) {
        const int row = tm + wm * 32 + fm * 16 + lg * 8 + r;
        const int col = tn + wn * 32 + fn * 16 + lr;
        const float v = acc[fm][fn][r] + bias[col];
        if (MODE == 0) {
          const int b = row / S, s = row % S;
          const int h = col >> 6, dk = col & 63;
          outh[(((size_t)(b * 16 + h) * S) + s) * 64 + dk] = (_Float16)v;
        } else {
          outf[(size_t)row * N + col] = v;
        }
      }
}

// ---------------------------------------------------------------------------
// Flash attention (online softmax). Grid: (S/64, H, B), 128 threads = 4 waves.
// Each wave owns 16 query rows; loop over keys in tiles of 32.
// K tile staged by TDM; V tile transposed manually (TDM cannot transpose).
// Per tile per wave: 4 WMMA (QK^T) + 4 WMMA (P*V).
// ---------------------------------------------------------------------------
__global__ __launch_bounds__(128) void mha_flash_wmma(
    const _Float16* __restrict__ Qh, const _Float16* __restrict__ Kh,
    const _Float16* __restrict__ Vh, const int* __restrict__ mask,
    _Float16* __restrict__ O) {
  constexpr int S = 2048, DK = 64, H = 16;
  __shared__ _Float16 sK[32][64];      // key tile, row-major (key, dk)
  __shared__ _Float16 sVt[64][32];     // value tile, transposed (dk, key)
  __shared__ _Float16 sP[4][16][32];   // per-wave P tile for layout shuffle

  const int tid = threadIdx.x;
  const int wave = tid >> 5, lane = tid & 31;
  const int lg = lane >> 4, lr = lane & 15;
  const int b = blockIdx.z, h = blockIdx.y;
  const int q0 = blockIdx.x * 64 + wave * 16;

  const size_t head = ((size_t)b * H + h) * S * DK;
  const _Float16* Qb = Qh + head;
  const _Float16* Kb = Kh + head;
  const _Float16* Vb = Vh + head;
#if MHA_HAVE_TDM
  const unsigned sK_lds = (unsigned)(unsigned long long)(void*)&sK[0][0];
#endif

  // Q fragments for rows q0..q0+15, held for whole kernel (K dims 0..31, 32..63)
  const v8h* pq = (const v8h*)(Qb + (size_t)(q0 + lr) * DK);
  const v16h aq0 = cat16(pq[lg], pq[2 + lg]);
  const v16h aq1 = cat16(pq[4 + lg], pq[6 + lg]);

  v8f o[4] = {};         // O accumulator 16x64
  float mrow[8], lrow[8];
#pragma unroll
  for (int r = 0; r < 8; ++r) { mrow[r] = -1e30f; lrow[r] = 0.f; }
  const float scale = 0.125f;  // 1/sqrt(64)

  for (int kt = 0; kt < S; kt += 32) {
#if MHA_HAVE_TDM
    // DMA the 32x64 K tile into LDS; overlaps with the manual V transpose below
    if (wave == 0)
      tdm_load_2d_f16(sK_lds, Kb + (size_t)kt * DK, DK, S, DK, 32, DK);
#endif
    {  // stage V tile transposed into LDS (and K tile on the fallback path)
      const int row = tid >> 2, seg = tid & 3;
#if !MHA_HAVE_TDM
      const v8h* sk = (const v8h*)(Kb + (size_t)(kt + row) * DK + seg * 16);
      v8h* dk = (v8h*)&sK[row][seg * 16];
      dk[0] = sk[0];
      dk[1] = sk[1];
#endif
      const v8h* sv = (const v8h*)(Vb + (size_t)(kt + row) * DK + seg * 16);
      const v8h v0 = sv[0], v1 = sv[1];
#pragma unroll
      for (int j = 0; j < 8; ++j) sVt[seg * 16 + j][row] = v0[j];
#pragma unroll
      for (int j = 0; j < 8; ++j) sVt[seg * 16 + 8 + j][row] = v1[j];
    }
#if MHA_HAVE_TDM
    if (wave == 0) __builtin_amdgcn_s_wait_tensorcnt(0);
#endif
    __syncthreads();

    // scores S = Q * K^T : two 16x16 frags over the 32-key tile
    v8f sc[2] = {};
#pragma unroll
    for (int nc = 0; nc < 2; ++nc) {
      const v8h* pk = (const v8h*)&sK[nc * 16 + lr][0];
      sc[nc] = wmma_f16(aq0, cat16(pk[lg], pk[2 + lg]), sc[nc]);
      sc[nc] = wmma_f16(aq1, cat16(pk[4 + lg], pk[6 + lg]), sc[nc]);
    }

    float mb[2];
#pragma unroll
    for (int nc = 0; nc < 2; ++nc)
      mb[nc] = (mask[(size_t)b * S + kt + nc * 16 + lr] == 0) ? -1e9f : 0.f;

    // online softmax over the 32 new columns (row reduce across 16 lanes)
#pragma unroll
    for (int r = 0; r < 8; ++r) {
      float s0 = sc[0][r] * scale + mb[0];
      float s1 = sc[1][r] * scale + mb[1];
      float mx = fmaxf(s0, s1);
#pragma unroll
      for (int off = 1; off < 16; off <<= 1)
        mx = fmaxf(mx, __shfl_xor(mx, off, 32));
      const float mnew = fmaxf(mrow[r], mx);
      const float alpha = __expf(mrow[r] - mnew);
      mrow[r] = mnew;
      const float p0 = __expf(s0 - mnew);
      const float p1 = __expf(s1 - mnew);
      float ps = p0 + p1;
#pragma unroll
      for (int off = 1; off < 16; off <<= 1) ps += __shfl_xor(ps, off, 32);
      lrow[r] = lrow[r] * alpha + ps;
#pragma unroll
      for (int d = 0; d < 4; ++d) o[d][r] *= alpha;
      sP[wave][lg * 8 + r][lr] = (_Float16)p0;
      sP[wave][lg * 8 + r][16 + lr] = (_Float16)p1;
    }
    __syncthreads();

    // O += P(16x32) * V(32x64): P re-read in A-fragment layout from LDS
    const v8h* pp = (const v8h*)&sP[wave][lr][0];
    const v16h ap = cat16(pp[lg], pp[2 + lg]);
#pragma unroll
    for (int d = 0; d < 4; ++d) {
      const v8h* pv = (const v8h*)&sVt[d * 16 + lr][0];
      o[d] = wmma_f16(ap, cat16(pv[lg], pv[2 + lg]), o[d]);
    }
    __syncthreads();
  }

  // normalize and write to concat layout [B, S, H*DK]
#pragma unroll
  for (int d = 0; d < 4; ++d)
#pragma unroll
    for (int r = 0; r < 8; ++r) {
      const float v = o[d][r] / lrow[r];
      O[((size_t)b * S + q0 + lg * 8 + r) * (H * DK) + h * DK + d * 16 + lr] =
          (_Float16)v;
    }
}

// ---------------------------------------------------------------------------
// Host launcher
// ---------------------------------------------------------------------------
extern "C" void kernel_launch(void* const* d_in, const int* in_sizes, int n_in,
                              void* d_out, int out_size, void* d_ws,
                              size_t ws_size, hipStream_t stream) {
  (void)in_sizes; (void)n_in; (void)out_size; (void)ws_size;
  const float* q   = (const float*)d_in[0];
  const float* k   = (const float*)d_in[1];
  const float* v   = (const float*)d_in[2];
  const int* mask  = (const int*)d_in[3];
  const float* Wq  = (const float*)d_in[4];
  const float* bq  = (const float*)d_in[5];
  const float* Wk  = (const float*)d_in[6];
  const float* bk  = (const float*)d_in[7];
  const float* Wv  = (const float*)d_in[8];
  const float* bv  = (const float*)d_in[9];
  const float* Wo  = (const float*)d_in[10];
  const float* bo  = (const float*)d_in[11];

  constexpr int B = 2, S = 2048, D = 1024, H = 16;
  constexpr size_t MB = 1024 * 1024;
  char* ws = (char*)d_ws;
  _Float16* qh = (_Float16*)(ws + 0 * MB);    // Q  [B,H,S,64]  8 MB
  _Float16* kh = (_Float16*)(ws + 8 * MB);    // K  [B,H,S,64]  8 MB
  _Float16* vh = (_Float16*)(ws + 16 * MB);   // V  [B,H,S,64]  8 MB
  _Float16* xq = (_Float16*)(ws + 24 * MB);   // query f16      8 MB
  _Float16* xk = (_Float16*)(ws + 32 * MB);
  _Float16* xv = (_Float16*)(ws + 40 * MB);
  _Float16* wq = (_Float16*)(ws + 48 * MB);   // weights f16    2 MB each
  _Float16* wk = (_Float16*)(ws + 50 * MB);
  _Float16* wv = (_Float16*)(ws + 52 * MB);
  _Float16* wo = (_Float16*)(ws + 54 * MB);
  _Float16* ao = (_Float16*)(ws + 56 * MB);   // attn out [B,S,1024] 8 MB

  const int nact = B * S * D;  // 4M elements
  const int nw = D * D;        // 1M elements
  const int tpb = 256;
  const int ga4 = (nact / 4 + tpb - 1) / tpb;
  const int gw4 = (nw / 4 + tpb - 1) / tpb;
  mha_cvt_f32_to_f16<<<ga4, tpb, 0, stream>>>(q, xq, nact / 4);
  mha_cvt_f32_to_f16<<<ga4, tpb, 0, stream>>>(k, xk, nact / 4);
  mha_cvt_f32_to_f16<<<ga4, tpb, 0, stream>>>(v, xv, nact / 4);
  mha_cvt_f32_to_f16<<<gw4, tpb, 0, stream>>>(Wq, wq, nw / 4);
  mha_cvt_f32_to_f16<<<gw4, tpb, 0, stream>>>(Wk, wk, nw / 4);
  mha_cvt_f32_to_f16<<<gw4, tpb, 0, stream>>>(Wv, wv, nw / 4);
  mha_cvt_f32_to_f16<<<gw4, tpb, 0, stream>>>(Wo, wo, nw / 4);

  dim3 gg((B * S) / 128, D / 64);  // (32, 16)
  mha_gemm_wmma<0><<<gg, 256, 0, stream>>>(xq, wq, bq, qh, nullptr, B * S, D, D, S);
  mha_gemm_wmma<0><<<gg, 256, 0, stream>>>(xk, wk, bk, kh, nullptr, B * S, D, D, S);
  mha_gemm_wmma<0><<<gg, 256, 0, stream>>>(xv, wv, bv, vh, nullptr, B * S, D, D, S);

  dim3 gf(S / 64, H, B);  // (32, 16, 2)
  mha_flash_wmma<<<gf, 128, 0, stream>>>(qh, kh, vh, mask, ao);

  mha_gemm_wmma<1><<<gg, 256, 0, stream>>>(ao, wo, bo, nullptr, (float*)d_out,
                                           B * S, D, D, S);
}
